// MHA_31808527794565
// MI455X (gfx1250) — compile-verified, matrix operand order checked by reading, not run on hardware
//
#include <hip/hip_runtime.h>

// ---------------------------------------------------------------------------
// MI455X (gfx1250) linear-attention pipeline.
// bf16 WMMA (16x16x32) + Tensor Data Mover double-buffered GEMM staging.
// B=4, L=4096, D=1024, H=16, M=64.
// ---------------------------------------------------------------------------

typedef __attribute__((ext_vector_type(16))) __bf16 v16bf;
typedef __attribute__((ext_vector_type(8)))  __bf16 v8bf;
typedef __attribute__((ext_vector_type(8)))  float  v8f;
typedef unsigned int u32x4 __attribute__((ext_vector_type(4)));
typedef int          i32x8 __attribute__((ext_vector_type(8)));
typedef int          i32x4 __attribute__((ext_vector_type(4)));

#if __has_builtin(__builtin_amdgcn_tensor_load_to_lds)
#define HAS_TDM 1
#else
#define HAS_TDM 0
#endif

constexpr int Bb  = 4;
constexpr int L   = 4096;
constexpr int H   = 16;
constexpr int Mh  = 64;
constexpr int ROWS = Bb * L;          // 16384
constexpr int KD  = 1024;             // GEMM K (D or H*M)
constexpr int ND  = 1024;             // GEMM N (H*M or D)

__device__ __forceinline__ unsigned short f2bf(float f) {
    unsigned u = __float_as_uint(f);
    u = u + 0x7FFFu + ((u >> 16) & 1u);           // round-to-nearest-even
    return (unsigned short)(u >> 16);
}
__device__ __forceinline__ unsigned pack_bf2(float a, float b) {
    unsigned ua = __float_as_uint(a), ub = __float_as_uint(b);
    ua = ua + 0x7FFFu + ((ua >> 16) & 1u);
    ub = ub + 0x7FFFu + ((ub >> 16) & 1u);
    return (ua >> 16) | (ub & 0xFFFF0000u);
}
__device__ __forceinline__ float bf_lo(unsigned u) { return __uint_as_float(u << 16); }
__device__ __forceinline__ float bf_hi(unsigned u) { return __uint_as_float(u & 0xFFFF0000u); }

// 16-element bf16 fragment from two 16-byte LDS reads (CDNA5 16-bit layout:
// VGPRs 0-3 = one contiguous 8-element K-group, VGPRs 4-7 = K+16 group).
__device__ __forceinline__ v16bf frag16(const unsigned short* p0,
                                        const unsigned short* p1) {
    v8bf lo = *reinterpret_cast<const v8bf*>(p0);
    v8bf hi = *reinterpret_cast<const v8bf*>(p1);
    return __builtin_shufflevector(lo, hi,
        0,1,2,3,4,5,6,7,8,9,10,11,12,13,14,15);
}

__device__ __forceinline__ v8f wmma_bf16(v16bf a, v16bf b, v8f c) {
    return __builtin_amdgcn_wmma_f32_16x16x32_bf16(
        false, a, false, b, (short)0, c, false, false);
}

// ---------------------------------------------------------------------------
// TDM: async-load a 128-row x 64-byte tile (bf16 128x32, row pitch KD*2 B)
// into LDS with a 16B pad after each 64B row -> LDS row stride 80 B, which
// matches the padded fragment layout used by the ds_load_b128 reads.
//   data_size=8B: tile_dim0=8, tile_dim1=128, tensor_dim0_stride=KD*2/8=256.
// Toolchain here exposes the 6-arg builtin:
//   (u32x4 g0, i32x8 g1, i32x4 g2, i32x4 g3, i32x8 extra, i32 cpol)
// ---------------------------------------------------------------------------
#if HAS_TDM
__device__ __forceinline__ void tdm_load_tile(unsigned lds_addr,
                                              const unsigned short* gsrc) {
    const unsigned long long ga = (unsigned long long)(uintptr_t)gsrc;
    u32x4 g0;
    g0[0] = 1u;                                  // count=1 (valid user D#)
    g0[1] = lds_addr;                            // LDS byte address
    g0[2] = (unsigned)ga;                        // global_addr[31:0]
    g0[3] = (unsigned)(ga >> 32) | (2u << 30);   // global_addr[56:32] | type=2
    i32x8 g1;
    g1[0] = (3 << 16)        // data_size = 8 bytes
          | (1 << 20)        // pad_enable
          | (3 << 22)        // pad_interval: 16 DWORDs (one 64B row)
          | (3 << 25);       // pad_amount:   4 DWORDs (16 B)
    g1[1] = 256 << 16;       // tensor_dim0 = 256 (8B units) [low16 in 63:48]
    g1[2] = 128 << 16;       // tensor_dim0 hi=0 | tensor_dim1 = 128
    g1[3] = 8 << 16;         // tensor_dim1 hi=0 | tile_dim0 = 8 (64 B)
    g1[4] = 128;             // tile_dim1 = 128 rows, tile_dim2 = 0
    g1[5] = 256;             // tensor_dim0_stride = 256 (8B units) low32
    g1[6] = 0;               // stride hi | tensor_dim1_stride lo (unused, 2D)
    g1[7] = 0;
    const i32x4 z4 = {0, 0, 0, 0};
    const i32x8 z8 = {0, 0, 0, 0, 0, 0, 0, 0};
    __builtin_amdgcn_tensor_load_to_lds(g0, g1, z4, z4, z8, 0);
}
#endif

// ---------------------------------------------------------------------------
// Prep 1: f32 -> bf16 elementwise (query/key/value), 8 elems/thread.
// ---------------------------------------------------------------------------
__global__ __launch_bounds__(256)
void cvt_bf16_kernel(const float* __restrict__ in, unsigned* __restrict__ out) {
    const size_t i = ((size_t)blockIdx.x * 256 + threadIdx.x) * 8;
    const float4 a = *reinterpret_cast<const float4*>(in + i);
    const float4 b = *reinterpret_cast<const float4*>(in + i + 4);
    uint4 p;
    p.x = pack_bf2(a.x, a.y);
    p.y = pack_bf2(a.z, a.w);
    p.z = pack_bf2(b.x, b.y);
    p.w = pack_bf2(b.z, b.w);
    *reinterpret_cast<uint4*>(out + i / 2) = p;
}

// ---------------------------------------------------------------------------
// Prep 2: W[K=1024][N=1024] f32 -> Wt[N][K] bf16 (32x32 LDS tiles).
// ---------------------------------------------------------------------------
__global__ __launch_bounds__(256)
void wT_bf16_kernel(const float* __restrict__ W, unsigned short* __restrict__ Wt) {
    __shared__ float tile[32][33];
    const int k0 = blockIdx.y * 32;
    const int n0 = blockIdx.x * 32;
    const int t  = threadIdx.x;
    const int r  = t >> 3;
    const int c  = (t & 7) * 4;
    {
        const float4 v = *reinterpret_cast<const float4*>(
            W + (size_t)(k0 + r) * ND + n0 + c);
        tile[r][c] = v.x; tile[r][c+1] = v.y; tile[r][c+2] = v.z; tile[r][c+3] = v.w;
    }
    __syncthreads();
    unsigned short* o = Wt + (size_t)(n0 + r) * KD + k0 + c;
    #pragma unroll
    for (int i = 0; i < 4; ++i) o[i] = f2bf(tile[c + i][r]);
}

// ---------------------------------------------------------------------------
// GEMM: C[16384,1024] = act(A_bf16 x Wt_bf16^T + b)
// Block tile 128x128, 8 waves of 32x64 (2 A-frags x 4 B-frags -> 8 WMMA/step).
// TDM double-buffers both LDS tiles (wave 0 issues, s_wait_tensorcnt gates).
// ---------------------------------------------------------------------------
constexpr int BM = 128, BN = 128, BK = 32;
constexpr int LDP = BK + 8;                   // 40 ushorts = 80 B row stride
constexpr int NS  = KD / BK;                  // 32 K-steps

template<bool RELU_EPS, bool OUT_F32>
__global__ __launch_bounds__(256)
void gemm_bias_act_kernel(const unsigned short* __restrict__ Abf,
                          const unsigned short* __restrict__ Wt,  // [N][K] bf16
                          const float* __restrict__ bias,
                          void*        __restrict__ Cout) {
#if HAS_TDM
    __shared__ unsigned short sA[2][BM][LDP];
    __shared__ unsigned short sB[2][BN][LDP];
#else
    __shared__ unsigned short sA[1][BM][LDP];
    __shared__ unsigned short sB[1][BN][LDP];
#endif

    const int row0 = blockIdx.y * BM;
    const int col0 = blockIdx.x * BN;
    const int t    = threadIdx.x;
    const int lane = t & 31;
    const int wave = t >> 5;
    const int m16  = lane & 15;
    const int kg   = lane >> 4;
    const int wr   = (wave & 3) * 32;
    const int wc   = (wave >> 2) * 64;

    v8f acc[2][4];
    #pragma unroll
    for (int i = 0; i < 2; ++i)
        #pragma unroll
        for (int n = 0; n < 4; ++n)
            #pragma unroll
            for (int j = 0; j < 8; ++j) acc[i][n][j] = 0.f;

#if HAS_TDM
    const unsigned short* Ag = Abf + (size_t)row0 * KD;
    const unsigned short* Bg = Wt  + (size_t)col0 * KD;
    const unsigned ldsA[2] = { (unsigned)(uintptr_t)&sA[0][0][0],
                               (unsigned)(uintptr_t)&sA[1][0][0] };
    const unsigned ldsB[2] = { (unsigned)(uintptr_t)&sB[0][0][0],
                               (unsigned)(uintptr_t)&sB[1][0][0] };
    if (wave == 0) {
        tdm_load_tile(ldsA[0], Ag);
        tdm_load_tile(ldsB[0], Bg);
        tdm_load_tile(ldsA[1], Ag + BK);
        tdm_load_tile(ldsB[1], Bg + BK);
    }
    for (int s = 0; s < NS; ++s) {
        const int buf = s & 1;
        if (wave == 0) {
            if (s + 1 < NS) __builtin_amdgcn_s_wait_tensorcnt(2);
            else            __builtin_amdgcn_s_wait_tensorcnt(0);
        }
        __syncthreads();                       // tile `buf` visible to all

        const unsigned short (*cA)[LDP] = sA[buf];
        const unsigned short (*cB)[LDP] = sB[buf];
        const int ar0 = wr + m16, ar1 = wr + 16 + m16;
        v16bf a0 = frag16(&cA[ar0][kg * 8], &cA[ar0][16 + kg * 8]);
        v16bf a1 = frag16(&cA[ar1][kg * 8], &cA[ar1][16 + kg * 8]);
        #pragma unroll
        for (int n = 0; n < 4; ++n) {
            const int bn = wc + n * 16 + m16;
            v16bf bf = frag16(&cB[bn][kg * 8], &cB[bn][16 + kg * 8]);
            acc[0][n] = wmma_bf16(a0, bf, acc[0][n]);
            acc[1][n] = wmma_bf16(a1, bf, acc[1][n]);
        }
        __syncthreads();                       // all done reading `buf`
        if (wave == 0 && s + 2 < NS) {
            tdm_load_tile(ldsA[buf], Ag + (size_t)(s + 2) * BK);
            tdm_load_tile(ldsB[buf], Bg + (size_t)(s + 2) * BK);
        }
    }
#else
    const int sr = t >> 1;
    const int sc = (t & 1) * 16;
    for (int kk = 0; kk < KD; kk += BK) {
        {
            const uint4* g = reinterpret_cast<const uint4*>(
                Abf + (size_t)(row0 + sr) * KD + kk + sc);
            *reinterpret_cast<uint4*>(&sA[0][sr][sc])     = g[0];
            *reinterpret_cast<uint4*>(&sA[0][sr][sc + 8]) = g[1];
        }
        {
            const uint4* g = reinterpret_cast<const uint4*>(
                Wt + (size_t)(col0 + sr) * KD + kk + sc);
            *reinterpret_cast<uint4*>(&sB[0][sr][sc])     = g[0];
            *reinterpret_cast<uint4*>(&sB[0][sr][sc + 8]) = g[1];
        }
        __syncthreads();
        const int ar0 = wr + m16, ar1 = wr + 16 + m16;
        v16bf a0 = frag16(&sA[0][ar0][kg * 8], &sA[0][ar0][16 + kg * 8]);
        v16bf a1 = frag16(&sA[0][ar1][kg * 8], &sA[0][ar1][16 + kg * 8]);
        #pragma unroll
        for (int n = 0; n < 4; ++n) {
            const int bn = wc + n * 16 + m16;
            v16bf bf = frag16(&sB[0][bn][kg * 8], &sB[0][bn][16 + kg * 8]);
            acc[0][n] = wmma_bf16(a0, bf, acc[0][n]);
            acc[1][n] = wmma_bf16(a1, bf, acc[1][n]);
        }
        __syncthreads();
    }
#endif

    #pragma unroll
    for (int i = 0; i < 2; ++i) {
        const int orow = row0 + wr + i * 16 + kg * 8;
        #pragma unroll
        for (int n = 0; n < 4; ++n) {
            const int ocol = col0 + wc + n * 16 + m16;
            const float bv = bias[ocol];
            #pragma unroll
            for (int j = 0; j < 8; ++j) {
                float v = acc[i][n][j] + bv;
                if (RELU_EPS) v = fmaxf(v, 0.f) + 1e-3f;
                const size_t idx = (size_t)(orow + j) * ND + ocol;
                if (OUT_F32) ((float*)Cout)[idx] = v;
                else         ((unsigned short*)Cout)[idx] = f2bf(v);
            }
        }
    }
}

// ---------------------------------------------------------------------------
// kvs[b,h][64,64] = sum_l k[l]^T v[l]   and   ks_sum[b,h][64].
// ---------------------------------------------------------------------------
constexpr int CH = 32;
constexpr int LDC = CH + 8;

__global__ __launch_bounds__(256)
void kv_state_kernel(const unsigned short* __restrict__ kbf,
                     const unsigned short* __restrict__ vbf,
                     float* __restrict__ kvs,
                     float* __restrict__ ksum) {
    __shared__ unsigned short kT[Mh][LDC];
    __shared__ unsigned short vT[Mh][LDC];

    const int bh   = blockIdx.x;
    const int b    = bh >> 4, h = bh & 15;
    const int t    = threadIdx.x;
    const int lane = t & 31, wave = t >> 5;
    const int m16  = lane & 15, kg = lane >> 4;
    const int tileR  = wave >> 1;
    const int tileC0 = (wave & 1) * 2;

    v8f acc0, acc1;
    #pragma unroll
    for (int j = 0; j < 8; ++j) { acc0[j] = 0.f; acc1[j] = 0.f; }
    float ks = 0.f;

    for (int l0 = 0; l0 < L; l0 += CH) {
        {
            const int l  = t >> 3;
            const int mm = (t & 7) * 8;
            const size_t base = (((size_t)b * L + l0 + l) * H + h) * Mh + mm;
            const uint4 kv4 = *reinterpret_cast<const uint4*>(kbf + base);
            const uint4 vv4 = *reinterpret_cast<const uint4*>(vbf + base);
            const unsigned short* kp = (const unsigned short*)&kv4;
            const unsigned short* vp = (const unsigned short*)&vv4;
            #pragma unroll
            for (int i = 0; i < 8; ++i) kT[mm + i][l] = kp[i];
            #pragma unroll
            for (int i = 0; i < 8; ++i) vT[mm + i][l] = vp[i];
        }
        __syncthreads();

        const int am = tileR * 16 + m16;
        v16bf afrag = frag16(&kT[am][kg * 8], &kT[am][16 + kg * 8]);
        const int bn0 = tileC0 * 16 + m16;
        v16bf b0 = frag16(&vT[bn0][kg * 8], &vT[bn0][16 + kg * 8]);
        acc0 = wmma_bf16(afrag, b0, acc0);
        const int bn1 = (tileC0 + 1) * 16 + m16;
        v16bf b1 = frag16(&vT[bn1][kg * 8], &vT[bn1][16 + kg * 8]);
        acc1 = wmma_bf16(afrag, b1, acc1);

        if (t < Mh) {
            const unsigned* row = reinterpret_cast<const unsigned*>(&kT[t][0]);
            #pragma unroll
            for (int i = 0; i < CH / 2; ++i) {
                const unsigned u = row[i];
                ks += bf_lo(u) + bf_hi(u);
            }
        }
        __syncthreads();
    }

    const int mrow = tileR * 16 + kg * 8;
    const size_t obase = (size_t)bh * Mh * Mh;
    #pragma unroll
    for (int j = 0; j < 8; ++j) {
        kvs[obase + (size_t)(mrow + j) * Mh + tileC0 * 16 + m16]       = acc0[j];
        kvs[obase + (size_t)(mrow + j) * Mh + (tileC0 + 1) * 16 + m16] = acc1[j];
    }
    if (t < Mh) ksum[(size_t)bh * Mh + t] = ks;
}

// ---------------------------------------------------------------------------
// a_vn[b,l,h,:] = (q . kvs) / (q . ks_sum); block = (b,h,128-row l-tile).
// ---------------------------------------------------------------------------
constexpr int TL  = 128;
constexpr int LDQ = Mh + 8;

__global__ __launch_bounds__(256)
void av_norm_kernel(const unsigned short* __restrict__ qbf,
                    const float* __restrict__ kvs,
                    const float* __restrict__ ksum,
                    unsigned short* __restrict__ avn) {
    __shared__ unsigned short sQ[TL][LDQ];
    __shared__ unsigned short sKV[Mh][LDQ];
    __shared__ float sKS[Mh];
    __shared__ float sNorm[TL];

    const int b  = blockIdx.z;
    const int h  = blockIdx.y;
    const int l0 = blockIdx.x * TL;
    const int t    = threadIdx.x;
    const int lane = t & 31, wave = t >> 5;
    const int m16  = lane & 15, kg = lane >> 4;

    {
        const int r = t >> 1, c = (t & 1) * 32;
        const uint4* g = reinterpret_cast<const uint4*>(
            qbf + (((size_t)b * L + l0 + r) * H + h) * Mh + c);
        *reinterpret_cast<uint4*>(&sQ[r][c])      = g[0];
        *reinterpret_cast<uint4*>(&sQ[r][c + 8])  = g[1];
        *reinterpret_cast<uint4*>(&sQ[r][c + 16]) = g[2];
        *reinterpret_cast<uint4*>(&sQ[r][c + 24]) = g[3];
    }
    {
        const int m = t >> 2, d0 = (t & 3) * 16;
        const size_t base = ((size_t)(b * H + h) * Mh + m) * Mh + d0;
        #pragma unroll
        for (int i = 0; i < 16; ++i) sKV[d0 + i][m] = f2bf(kvs[base + i]);
    }
    if (t < Mh) sKS[t] = ksum[(size_t)(b * H + h) * Mh + t];
    __syncthreads();

    if (t < TL) {
        float s = 0.f;
        const unsigned* row = reinterpret_cast<const unsigned*>(&sQ[t][0]);
        #pragma unroll
        for (int m = 0; m < Mh / 2; ++m) {
            const unsigned u = row[m];
            s += bf_lo(u) * sKS[2 * m] + bf_hi(u) * sKS[2 * m + 1];
        }
        sNorm[t] = s;
    }

    v8f acc[4];
    #pragma unroll
    for (int i = 0; i < 4; ++i)
        #pragma unroll
        for (int j = 0; j < 8; ++j) acc[i][j] = 0.f;

    #pragma unroll
    for (int kk = 0; kk < Mh; kk += 32) {
        const int ar = wave * 16 + m16;
        v16bf afrag = frag16(&sQ[ar][kk + kg * 8], &sQ[ar][kk + 16 + kg * 8]);
        #pragma unroll
        for (int n = 0; n < 4; ++n) {
            const int bn = n * 16 + m16;
            v16bf bfrag = frag16(&sKV[bn][kk + kg * 8], &sKV[bn][kk + 16 + kg * 8]);
            acc[n] = wmma_bf16(afrag, bfrag, acc[n]);
        }
    }
    __syncthreads();

    const int rl = wave * 16 + kg * 8;
    #pragma unroll
    for (int n = 0; n < 4; ++n) {
        const int d = n * 16 + m16;
        #pragma unroll
        for (int j = 0; j < 8; ++j) {
            const int row = rl + j;
            const float v = acc[n][j] / sNorm[row];
            avn[(((size_t)b * L + l0 + row) * H + h) * Mh + d] = f2bf(v);
        }
    }
}

// ---------------------------------------------------------------------------
// Workspace layout (~201 MB): xq/xk/xv bf16 inputs (xq reused as a_vn),
// q/k/v bf16 projections, 4x transposed bf16 weights, kvs + ks_sum f32.
// ---------------------------------------------------------------------------
extern "C" void kernel_launch(void* const* d_in, const int* in_sizes, int n_in,
                              void* d_out, int out_size, void* d_ws, size_t ws_size,
                              hipStream_t stream) {
    const float* query = (const float*)d_in[0];
    const float* key   = (const float*)d_in[1];
    const float* value = (const float*)d_in[2];
    const float* Wq    = (const float*)d_in[3];
    const float* bq    = (const float*)d_in[4];
    const float* Wk    = (const float*)d_in[5];
    const float* bk    = (const float*)d_in[6];
    const float* Wv    = (const float*)d_in[7];
    const float* bv    = (const float*)d_in[8];
    const float* Wo    = (const float*)d_in[9];
    const float* bo    = (const float*)d_in[10];

    const size_t TEN_BYTES = (size_t)ROWS * ND * sizeof(unsigned short); // 32 MB
    const size_t WT_BYTES  = (size_t)KD * ND * sizeof(unsigned short);   //  2 MB
    const size_t KVS_BYTES = (size_t)Bb * H * Mh * Mh * sizeof(float);   //  1 MB

    unsigned char* ws = (unsigned char*)d_ws;
    unsigned short* xq  = (unsigned short*)(ws);
    unsigned short* xk  = (unsigned short*)(ws + TEN_BYTES);
    unsigned short* xv  = (unsigned short*)(ws + 2 * TEN_BYTES);
    unsigned short* qb  = (unsigned short*)(ws + 3 * TEN_BYTES);
    unsigned short* kb  = (unsigned short*)(ws + 4 * TEN_BYTES);
    unsigned short* vb  = (unsigned short*)(ws + 5 * TEN_BYTES);
    unsigned short* wqT = (unsigned short*)(ws + 6 * TEN_BYTES);
    unsigned short* wkT = (unsigned short*)(ws + 6 * TEN_BYTES + WT_BYTES);
    unsigned short* wvT = (unsigned short*)(ws + 6 * TEN_BYTES + 2 * WT_BYTES);
    unsigned short* woT = (unsigned short*)(ws + 6 * TEN_BYTES + 3 * WT_BYTES);
    float* kvs = (float*)(ws + 6 * TEN_BYTES + 4 * WT_BYTES);
    float* ksm = (float*)(ws + 6 * TEN_BYTES + 4 * WT_BYTES + KVS_BYTES);
    unsigned short* avn = xq;     // xq dead after q projection

    const int nCvtBlocks = (int)(((size_t)ROWS * KD) / (256 * 8));     // 8192
    cvt_bf16_kernel<<<nCvtBlocks, 256, 0, stream>>>(query, (unsigned*)xq);
    cvt_bf16_kernel<<<nCvtBlocks, 256, 0, stream>>>(key,   (unsigned*)xk);
    cvt_bf16_kernel<<<nCvtBlocks, 256, 0, stream>>>(value, (unsigned*)xv);
    const dim3 gt(ND / 32, KD / 32);
    wT_bf16_kernel<<<gt, 256, 0, stream>>>(Wq, wqT);
    wT_bf16_kernel<<<gt, 256, 0, stream>>>(Wk, wkT);
    wT_bf16_kernel<<<gt, 256, 0, stream>>>(Wv, wvT);
    wT_bf16_kernel<<<gt, 256, 0, stream>>>(Wo, woT);

    const dim3 gg(ND / BN, ROWS / BM);                                 // (8,128)
    gemm_bias_act_kernel<true,  false><<<gg, 256, 0, stream>>>(xq, wqT, bq, qb);
    gemm_bias_act_kernel<true,  false><<<gg, 256, 0, stream>>>(xk, wkT, bk, kb);
    gemm_bias_act_kernel<false, false><<<gg, 256, 0, stream>>>(xv, wvT, bv, vb);

    kv_state_kernel<<<Bb * H, 256, 0, stream>>>(kb, vb, kvs, ksm);
    av_norm_kernel<<<dim3(L / TL, H, Bb), 256, 0, stream>>>(qb, kvs, ksm, avn);

    gemm_bias_act_kernel<false, true><<<gg, 256, 0, stream>>>(avn, woT, bo, (float*)d_out);
}
